// Block_Base_34832184771058
// MI455X (gfx1250) — matrix operand order repaired and checked by reading, not compiled, not run on hardware
//
#include <hip/hip_runtime.h>
#include <cstdint>
#include <cstddef>

// ---------------------------------------------------------------------------
// CDNA5 (gfx1250) transformer block: bf16 WMMA GEMMs + flash attention.
// Matrix math: v_wmma_f32_16x16x32_bf16 (fp32 accumulation).
// GEMM tile staging: GLOBAL_LOAD_ASYNC_TO_LDS_B128 + ASYNCcnt double buffering.
// ---------------------------------------------------------------------------

typedef __attribute__((ext_vector_type(16))) __bf16 v16bf;
typedef __attribute__((ext_vector_type(8)))  float  v8f;
typedef int v4i __attribute__((vector_size(16)));   // matches async-LDS builtin param type

union FragBf { uint4 u[2]; v16bf v; };   // 32 bytes = 16 bf16 = one WMMA A/B operand

__device__ __forceinline__ __bf16 f2bf(float f) {
  uint32_t x = __builtin_bit_cast(uint32_t, f);
  uint32_t r = x + 0x7FFFu + ((x >> 16) & 1u);   // round-to-nearest-even
  uint16_t h = (uint16_t)(r >> 16);
  return __builtin_bit_cast(__bf16, h);
}
__device__ __forceinline__ float bf2f(__bf16 b) {
  uint16_t h = __builtin_bit_cast(uint16_t, b);
  return __builtin_bit_cast(float, (uint32_t)h << 16);
}
__device__ __forceinline__ v8f vzero8() {
  v8f z = {0.f,0.f,0.f,0.f,0.f,0.f,0.f,0.f};
  return z;
}

// --- CDNA5 async global->LDS copy (ASYNCcnt path), with safe fallback -------
#if defined(__has_builtin)
#if __has_builtin(__builtin_amdgcn_global_load_async_to_lds_b128)
#define HAS_ASYNC_LDS 1
#endif
#if __has_builtin(__builtin_amdgcn_s_wait_asynccnt)
#define HAS_WAIT_ASYNC 1
#endif
#endif

__device__ __forceinline__ void async_copy_b128(const void* gsrc, void* lds_dst) {
#ifdef HAS_ASYNC_LDS
  // param 1: global-AS v4i* (non-const); param 2: LDS-AS v4i*.
  // generic LDS pointer low 32 bits == LDS byte offset (ISA 10.2 aperture rules)
  __attribute__((address_space(1))) v4i* g =
      (__attribute__((address_space(1))) v4i*)(uintptr_t)gsrc;
  __attribute__((address_space(3))) v4i* l =
      (__attribute__((address_space(3))) v4i*)(uintptr_t)(uint32_t)(uintptr_t)lds_dst;
  __builtin_amdgcn_global_load_async_to_lds_b128(g, l, 0, 0);
#else
  *(uint4*)lds_dst = *(const uint4*)gsrc;
#endif
}

template <int N>
__device__ __forceinline__ void wait_async() {
#ifdef HAS_WAIT_ASYNC
  __builtin_amdgcn_s_wait_asynccnt(N);
#else
  asm volatile("s_wait_asynccnt %0" :: "n"(N) : "memory");
#endif
}

// ---------------------------------------------------------------------------
// Weight fp32 [K][N] -> bf16 transposed [N][K]
// ---------------------------------------------------------------------------
__global__ __launch_bounds__(256) void transpose_cast_kernel(
    const float* __restrict__ W, __bf16* __restrict__ Wt, int K, int N) {
  __shared__ float tile[32][33];
  const int n0 = blockIdx.x * 32, k0 = blockIdx.y * 32;
  const int tx = threadIdx.x, ty = threadIdx.y;          // block (32,8)
  #pragma unroll
  for (int i = 0; i < 32; i += 8)
    tile[ty + i][tx] = W[(size_t)(k0 + ty + i) * N + (n0 + tx)];
  __syncthreads();
  #pragma unroll
  for (int i = 0; i < 32; i += 8)
    Wt[(size_t)(n0 + ty + i) * K + (k0 + tx)] = f2bf(tile[tx][ty + i]);
}

// ---------------------------------------------------------------------------
// RMSNorm: out_bf16[row] = x * rsqrt(mean(x^2)+eps) * g
// ---------------------------------------------------------------------------
__global__ __launch_bounds__(256) void rmsnorm_kernel(
    const float* __restrict__ X, const float* __restrict__ g,
    __bf16* __restrict__ out, int C) {
  __shared__ float red[256];
  const int row = blockIdx.x;
  const float* xr = X + (size_t)row * C;
  float s = 0.f;
  for (int c = threadIdx.x; c < C; c += 256) { float v = xr[c]; s += v * v; }
  red[threadIdx.x] = s; __syncthreads();
  for (int st = 128; st > 0; st >>= 1) {
    if (threadIdx.x < st) red[threadIdx.x] += red[threadIdx.x + st];
    __syncthreads();
  }
  const float r = rsqrtf(red[0] / (float)C + 1e-6f);
  for (int c = threadIdx.x; c < C; c += 256)
    out[(size_t)row * C + c] = f2bf(xr[c] * r * g[c]);
}

// ---------------------------------------------------------------------------
// WMMA GEMM: C[M,N] = A[M,K](bf16) @ Bt[N,K](bf16)^T, fp32 accum.
// 256 threads, 128x128 tile, 8 waves (2x4), 64x32 per wave, K-step 32.
// LDS double buffer fed by async global->LDS B128 copies (4 per thread/tile).
// Epilogues (templated):
//  1: silu(acc+b) -> bf16     2: acc+b -> bf16 [B,H,T,D]   3: acc+b -> bf16 [B,H,D,T]
//  4: gelu(acc+b) -> bf16     5: (acc+b)*gate -> bf16      6: acc+b+resid -> fp32
// ---------------------------------------------------------------------------
template <int EPI>
__global__ __launch_bounds__(256, 1) void gemm_bf16_wmma(
    const __bf16* __restrict__ A, const __bf16* __restrict__ Bt,
    const float* __restrict__ bias, const float* __restrict__ resid,
    const __bf16* __restrict__ gate, __bf16* __restrict__ outb,
    float* __restrict__ outf, int M, int N, int K, int T, int H, int D) {
  __shared__ __attribute__((aligned(16))) __bf16 As[2][128][40];  // 80B rows: b128-aligned
  __shared__ __attribute__((aligned(16))) __bf16 Bs[2][128][40];

  const int tid  = threadIdx.x;
  const int lane = tid & 31, wave = tid >> 5;
  const int wm = wave >> 2, wn = wave & 3;              // 2 x 4 wave grid
  const int half = lane >> 4, l16 = lane & 15;
  const int m0 = blockIdx.y * 128, n0 = blockIdx.x * 128;

  // per-thread copy slots: 2 rows of A + 2 rows of B per tile
  const int crow0 = tid >> 2, ccol = (tid & 3) * 8;
  const int crow1 = (tid + 256) >> 2;

  auto issue_tile = [&](int k0, int buf) {
    async_copy_b128(&A [(size_t)(m0 + crow0) * K + k0 + ccol], &As[buf][crow0][ccol]);
    async_copy_b128(&Bt[(size_t)(n0 + crow0) * K + k0 + ccol], &Bs[buf][crow0][ccol]);
    async_copy_b128(&A [(size_t)(m0 + crow1) * K + k0 + ccol], &As[buf][crow1][ccol]);
    async_copy_b128(&Bt[(size_t)(n0 + crow1) * K + k0 + ccol], &Bs[buf][crow1][ccol]);
  };

  v8f acc[4][2];
  #pragma unroll
  for (int i = 0; i < 4; ++i)
    #pragma unroll
    for (int j = 0; j < 2; ++j) acc[i][j] = vzero8();

  const int nk = K >> 5;
  issue_tile(0, 0);

  for (int kt = 0; kt < nk; ++kt) {
    const int buf = kt & 1;
    if (kt + 1 < nk) {
      issue_tile((kt + 1) << 5, buf ^ 1);
      wait_async<4>();          // tile-kt copies done (async loads complete in order)
    } else {
      wait_async<0>();
    }
    __syncthreads();

    FragBf a[4], b[2];
    #pragma unroll
    for (int i = 0; i < 4; ++i) {                       // A frag: row=l16, K by half
      const int row = wm * 64 + i * 16 + l16;
      a[i].u[0] = *(const uint4*)&As[buf][row][half * 8];
      a[i].u[1] = *(const uint4*)&As[buf][row][16 + half * 8];
    }
    #pragma unroll
    for (int j = 0; j < 2; ++j) {                       // B frag: col=l16, K by half
      const int row = wn * 32 + j * 16 + l16;
      b[j].u[0] = *(const uint4*)&Bs[buf][row][half * 8];
      b[j].u[1] = *(const uint4*)&Bs[buf][row][16 + half * 8];
    }
    #pragma unroll
    for (int i = 0; i < 4; ++i)
      #pragma unroll
      for (int j = 0; j < 2; ++j)
        acc[i][j] = __builtin_amdgcn_wmma_f32_16x16x32_bf16(
            false, a[i].v, false, b[j].v, (short)0, acc[i][j], false, false);
    __syncthreads();            // compute done before tile kt+2 overwrites this buffer
  }

  // Epilogue: C/D layout -> lane l: row = r + 8*half, col = l16
  #pragma unroll
  for (int i = 0; i < 4; ++i) {
    #pragma unroll
    for (int j = 0; j < 2; ++j) {
      const int mt = m0 + wm * 64 + i * 16;
      const int nt = n0 + wn * 32 + j * 16;
      #pragma unroll
      for (int r = 0; r < 8; ++r) {
        const int m = mt + r + 8 * half;
        const int n = nt + l16;
        float v = acc[i][j][r] + bias[n];
        const size_t lin = (size_t)m * N + n;
        if (EPI == 1) {
          outb[lin] = f2bf(v / (1.f + __expf(-v)));
        } else if (EPI == 2) {
          const int bb = m / T, t = m % T, h = n / D, d = n % D;
          outb[(size_t)(((bb * H + h) * T + t)) * D + d] = f2bf(v);
        } else if (EPI == 3) {
          const int bb = m / T, t = m % T, h = n / D, d = n % D;
          outb[(size_t)(((bb * H + h) * D + d)) * T + t] = f2bf(v);
        } else if (EPI == 4) {
          outb[lin] = f2bf(0.5f * v * (1.f + erff(v * 0.70710678118654752f)));
        } else if (EPI == 5) {
          outb[lin] = f2bf(v * bf2f(gate[lin]));
        } else if (EPI == 6) {
          outf[lin] = v + resid[lin];
        } else {
          outb[lin] = f2bf(v);
        }
      }
    }
  }
}

// ---------------------------------------------------------------------------
// Flash attention with block-causal mask: (j mod 512) <= (i mod 512).
// Q,K in [B,H,T,64] bf16; V transposed [B,H,64,T] bf16; Y out [B,T,C] bf16.
// One wave per 16 query rows; 32 keys per step: 4 WMMA (S) + 4 WMMA (PV).
// ---------------------------------------------------------------------------
__global__ __launch_bounds__(256, 1) void attn_kernel(
    const __bf16* __restrict__ Q, const __bf16* __restrict__ Kb,
    const __bf16* __restrict__ Vt, __bf16* __restrict__ Y, int H, int T) {
  constexpr int D = 64;
  constexpr float scale = 0.125f;                      // 1/sqrt(64)
  __shared__ __attribute__((aligned(16))) __bf16 Pb[8][16][32];  // per-wave P bounce

  const int tid  = threadIdx.x;
  const int lane = tid & 31, wave = tid >> 5;
  const int half = lane >> 4, l16 = lane & 15;
  const int bh = blockIdx.y, b = bh / H, h = bh % H;
  const int qbase = blockIdx.x * 128 + wave * 16;
  const int qb512 = qbase & 511;

  const __bf16* Qp = Q  + (size_t)(b * H + h) * T * D;
  const __bf16* Kp = Kb + (size_t)(b * H + h) * T * D;
  const __bf16* Vp = Vt + (size_t)(b * H + h) * D * T;

  FragBf a0, a1;                                       // Q rows as A fragments (d=0..31, 32..63)
  {
    const __bf16* qr = Qp + (size_t)(qbase + l16) * D;
    a0.u[0] = *(const uint4*)(qr + half * 8);
    a0.u[1] = *(const uint4*)(qr + 16 + half * 8);
    a1.u[0] = *(const uint4*)(qr + 32 + half * 8);
    a1.u[1] = *(const uint4*)(qr + 48 + half * 8);
  }

  float mstate[8], lstate[8];
  v8f O[4];
  #pragma unroll
  for (int r = 0; r < 8; ++r) { mstate[r] = -3e38f; lstate[r] = 0.f; }
  #pragma unroll
  for (int f = 0; f < 4; ++f) O[f] = vzero8();

  for (int jb = 0; jb < T; jb += 32) {
    const int jb512 = jb & 511;
    if (jb512 > qb512 + 15) continue;                  // tile fully masked (wave-uniform)
    const bool full = (jb512 + 31) <= qb512;           // no element masking needed

    FragBf k00, k01, k10, k11;
    {
      const __bf16* kr0 = Kp + (size_t)(jb + l16) * D + half * 8;
      const __bf16* kr1 = Kp + (size_t)(jb + 16 + l16) * D + half * 8;
      k00.u[0] = *(const uint4*)(kr0);      k00.u[1] = *(const uint4*)(kr0 + 16);
      k01.u[0] = *(const uint4*)(kr0 + 32); k01.u[1] = *(const uint4*)(kr0 + 48);
      k10.u[0] = *(const uint4*)(kr1);      k10.u[1] = *(const uint4*)(kr1 + 16);
      k11.u[0] = *(const uint4*)(kr1 + 32); k11.u[1] = *(const uint4*)(kr1 + 48);
    }
    v8f S0 = vzero8(), S1 = vzero8();
    S0 = __builtin_amdgcn_wmma_f32_16x16x32_bf16(false, a0.v, false, k00.v, (short)0, S0, false, false);
    S0 = __builtin_amdgcn_wmma_f32_16x16x32_bf16(false, a1.v, false, k01.v, (short)0, S0, false, false);
    S1 = __builtin_amdgcn_wmma_f32_16x16x32_bf16(false, a0.v, false, k10.v, (short)0, S1, false, false);
    S1 = __builtin_amdgcn_wmma_f32_16x16x32_bf16(false, a1.v, false, k11.v, (short)0, S1, false, false);

    // online softmax across the 32 keys; rows live across 16 lanes of each half
    #pragma unroll
    for (int r = 0; r < 8; ++r) {
      float s0 = S0[r] * scale, s1 = S1[r] * scale;
      if (!full) {
        const int rm = qb512 + r + 8 * half;
        if (jb512 + l16 > rm)      s0 = -3e38f;
        if (jb512 + 16 + l16 > rm) s1 = -3e38f;
      }
      float mx = fmaxf(s0, s1);
      #pragma unroll
      for (int off = 8; off >= 1; off >>= 1) mx = fmaxf(mx, __shfl_xor(mx, off));
      const float nm = fmaxf(mstate[r], mx);
      const float al = __expf(mstate[r] - nm);
      const float p0 = __expf(s0 - nm), p1 = __expf(s1 - nm);
      float sm = p0 + p1;
      #pragma unroll
      for (int off = 8; off >= 1; off >>= 1) sm += __shfl_xor(sm, off);
      lstate[r] = lstate[r] * al + sm;
      mstate[r] = nm;
      #pragma unroll
      for (int f = 0; f < 4; ++f) O[f][r] *= al;
      Pb[wave][r + 8 * half][l16]      = f2bf(p0);     // P in C layout -> LDS
      Pb[wave][r + 8 * half][l16 + 16] = f2bf(p1);
    }

    // re-read P in A layout (same-wave DS ops are in-order; compiler inserts dscnt waits)
    FragBf pA;
    {
      const __bf16* pr = &Pb[wave][l16][0];
      pA.u[0] = *(const uint4*)(pr + half * 8);
      pA.u[1] = *(const uint4*)(pr + 16 + half * 8);
    }
    #pragma unroll
    for (int f = 0; f < 4; ++f) {                      // O += P(16x32) @ V(32x16)
      const __bf16* vr = Vp + (size_t)(f * 16 + l16) * T + jb + half * 8;
      FragBf vB;
      vB.u[0] = *(const uint4*)(vr);
      vB.u[1] = *(const uint4*)(vr + 16);
      O[f] = __builtin_amdgcn_wmma_f32_16x16x32_bf16(
          false, pA.v, false, vB.v, (short)0, O[f], false, false);
    }
  }

  const int C = H * D;
  #pragma unroll
  for (int r = 0; r < 8; ++r) {
    const float inv = 1.f / lstate[r];
    const int i = qbase + r + 8 * half;
    #pragma unroll
    for (int f = 0; f < 4; ++f)
      Y[(size_t)(b * T + i) * C + (h * D + f * 16 + l16)] = f2bf(O[f][r] * inv);
  }
}

// ---------------------------------------------------------------------------
// Host-side orchestration
// ---------------------------------------------------------------------------
extern "C" void kernel_launch(void* const* d_in, const int* in_sizes, int n_in,
                              void* d_out, int out_size, void* d_ws, size_t ws_size,
                              hipStream_t stream) {
  (void)in_sizes; (void)n_in; (void)out_size; (void)ws_size;
  constexpr int B = 4, T = 1536, C = 768, H = 12, D = 64;
  constexpr int M = B * T, C4 = 4 * C;

  const float* x     = (const float*)d_in[0];
  const float* g1    = (const float*)d_in[1];
  const float* g2    = (const float*)d_in[2];
  const float* W_in  = (const float*)d_in[3];  const float* b_in  = (const float*)d_in[4];
  const float* W_act = (const float*)d_in[5];  const float* b_act = (const float*)d_in[6];
  const float* W_out = (const float*)d_in[7];  const float* b_out = (const float*)d_in[8];
  const float* Wq    = (const float*)d_in[9];  const float* bq    = (const float*)d_in[10];
  const float* Wk    = (const float*)d_in[11]; const float* bk    = (const float*)d_in[12];
  const float* Wv    = (const float*)d_in[13]; const float* bv    = (const float*)d_in[14];
  const float* Wp    = (const float*)d_in[15]; const float* bp    = (const float*)d_in[16];
  const float* W1    = (const float*)d_in[17]; const float* b1    = (const float*)d_in[18];
  const float* W2    = (const float*)d_in[19]; const float* b2    = (const float*)d_in[20];
  float* out = (float*)d_out;

  size_t off = 0;
  auto alloc = [&](size_t bytes) -> void* {
    void* p = (char*)d_ws + off;
    off = (off + bytes + 255) & ~(size_t)255;
    return p;
  };
  __bf16* wt_in  = (__bf16*)alloc((size_t)C * C * 2);
  __bf16* wt_act = (__bf16*)alloc((size_t)C * C * 2);
  __bf16* wt_out = (__bf16*)alloc((size_t)C * C * 2);
  __bf16* wt_q   = (__bf16*)alloc((size_t)C * C * 2);
  __bf16* wt_k   = (__bf16*)alloc((size_t)C * C * 2);
  __bf16* wt_v   = (__bf16*)alloc((size_t)C * C * 2);
  __bf16* wt_p   = (__bf16*)alloc((size_t)C * C * 2);
  __bf16* wt_1   = (__bf16*)alloc((size_t)C * C4 * 2);
  __bf16* wt_2   = (__bf16*)alloc((size_t)C4 * C * 2);
  __bf16* hb     = (__bf16*)alloc((size_t)M * C * 2);   // rmsnorm1 out; reused as Y
  __bf16* actb   = (__bf16*)alloc((size_t)M * C * 2);   // silu(h@W_act+b)
  __bf16* ub     = (__bf16*)alloc((size_t)M * C * 2);   // silu(h@W_in+b); reused as Z
  __bf16* Qb     = (__bf16*)alloc((size_t)M * C * 2);   // [B,H,T,D]; reused as m
  __bf16* Kbb    = (__bf16*)alloc((size_t)M * C * 2);   // [B,H,T,D]
  __bf16* Vtb    = (__bf16*)alloc((size_t)M * C * 2);   // [B,H,D,T]
  float*  X2     = (float*) alloc((size_t)M * C * 4);   // post-attn residual (fp32)
  __bf16* Gb     = (__bf16*)alloc((size_t)M * C4 * 2);  // gelu(m@W1+b1)
  __bf16* Yb = hb;
  __bf16* Zb = ub;
  __bf16* Mb = Qb;

  const dim3 tb32(32, 8);
  transpose_cast_kernel<<<dim3(C / 32,  C / 32),  tb32, 0, stream>>>(W_in,  wt_in,  C,  C);
  transpose_cast_kernel<<<dim3(C / 32,  C / 32),  tb32, 0, stream>>>(W_act, wt_act, C,  C);
  transpose_cast_kernel<<<dim3(C / 32,  C / 32),  tb32, 0, stream>>>(W_out, wt_out, C,  C);
  transpose_cast_kernel<<<dim3(C / 32,  C / 32),  tb32, 0, stream>>>(Wq,    wt_q,   C,  C);
  transpose_cast_kernel<<<dim3(C / 32,  C / 32),  tb32, 0, stream>>>(Wk,    wt_k,   C,  C);
  transpose_cast_kernel<<<dim3(C / 32,  C / 32),  tb32, 0, stream>>>(Wv,    wt_v,   C,  C);
  transpose_cast_kernel<<<dim3(C / 32,  C / 32),  tb32, 0, stream>>>(Wp,    wt_p,   C,  C);
  transpose_cast_kernel<<<dim3(C4 / 32, C / 32),  tb32, 0, stream>>>(W1,    wt_1,   C,  C4);
  transpose_cast_kernel<<<dim3(C / 32,  C4 / 32), tb32, 0, stream>>>(W2,    wt_2,   C4, C);

  rmsnorm_kernel<<<M, 256, 0, stream>>>(x, g1, hb, C);

  const dim3 gCC(C / 128, M / 128), gC4(C4 / 128, M / 128);
  gemm_bf16_wmma<1><<<gCC, 256, 0, stream>>>(hb, wt_act, b_act, nullptr, nullptr, actb, nullptr, M, C, C, T, H, D);
  gemm_bf16_wmma<1><<<gCC, 256, 0, stream>>>(hb, wt_in,  b_in,  nullptr, nullptr, ub,   nullptr, M, C, C, T, H, D);
  gemm_bf16_wmma<2><<<gCC, 256, 0, stream>>>(ub, wt_q, bq, nullptr, nullptr, Qb,  nullptr, M, C, C, T, H, D);
  gemm_bf16_wmma<2><<<gCC, 256, 0, stream>>>(ub, wt_k, bk, nullptr, nullptr, Kbb, nullptr, M, C, C, T, H, D);
  gemm_bf16_wmma<3><<<gCC, 256, 0, stream>>>(ub, wt_v, bv, nullptr, nullptr, Vtb, nullptr, M, C, C, T, H, D);

  attn_kernel<<<dim3(T / 128, B * H), 256, 0, stream>>>(Qb, Kbb, Vtb, Yb, H, T);

  gemm_bf16_wmma<5><<<gCC, 256, 0, stream>>>(Yb, wt_p,   bp,    nullptr, actb, Zb, nullptr, M, C, C, T, H, D);
  gemm_bf16_wmma<6><<<gCC, 256, 0, stream>>>(Zb, wt_out, b_out, x, nullptr, nullptr, X2,    M, C, C, T, H, D);

  rmsnorm_kernel<<<M, 256, 0, stream>>>(X2, g2, Mb, C);

  gemm_bf16_wmma<4><<<gC4, 256, 0, stream>>>(Mb, wt_1, b1, nullptr, nullptr, Gb, nullptr, M, C4, C,  T, H, D);
  gemm_bf16_wmma<6><<<gCC, 256, 0, stream>>>(Gb, wt_2, b2, X2, nullptr, nullptr, out,     M, C, C4, T, H, D);
}
